// LPSAGE_26113401160265
// MI455X (gfx1250) — compile-verified, matrix operand order checked by reading, not run on hardware
//
#include <hip/hip_runtime.h>
#include <hip/hip_bf16.h>

typedef __attribute__((ext_vector_type(16))) __bf16 v16bf;
typedef __attribute__((ext_vector_type(8)))  float  v8f;
typedef unsigned short ushort_t;

// ---------------------------------------------------------------- helpers
__device__ __forceinline__ ushort_t f32_to_bf16(float f) {
    unsigned u = __builtin_bit_cast(unsigned, f);
    unsigned r = u + 0x7FFFu + ((u >> 16) & 1u);   // round-to-nearest-even
    return (ushort_t)(r >> 16);
}

// One bf16 WMMA fragment: 8 bf16 at base, 8 bf16 at base+16 elems (2x b128).
__device__ __forceinline__ v16bf load_frag(const ushort_t* base) {
    union { v16bf v; uint4 q[2]; } u;
    u.q[0] = *reinterpret_cast<const uint4*>(base);
    u.q[1] = *reinterpret_cast<const uint4*>(base + 16);
    return u.v;
}

__device__ __forceinline__ v8f wmma_bf16(v16bf a, v16bf b, v8f c) {
    return __builtin_amdgcn_wmma_f32_16x16x32_bf16(false, a, false, b,
                                                   (short)0, c, false, false);
}

// ---------------------------------------------------------------- utility
__global__ void fill_zero_f32(float* p, int n) {
    int i = blockIdx.x * blockDim.x + threadIdx.x;
    if (i < n) p[i] = 0.0f;
}
__global__ void fill_zero_i32(int* p, int n) {
    int i = blockIdx.x * blockDim.x + threadIdx.x;
    if (i < n) p[i] = 0;
}

// ---------------------------------------------------------------- CSR build
__global__ void deg_count(const int* __restrict__ dst, int* __restrict__ deg, int E) {
    int e = blockIdx.x * blockDim.x + threadIdx.x;
    if (e < E) atomicAdd(&deg[dst[e]], 1);
}

// Per-block inclusive scan (256 elems) -> exclusive ptr + block totals.
__global__ void scan_block(const int* __restrict__ deg, int* __restrict__ ptr,
                           int* __restrict__ blockSums, int N) {
    __shared__ int s[256];
    int t = threadIdx.x;
    int i = blockIdx.x * 256 + t;
    int v = (i < N) ? deg[i] : 0;
    s[t] = v;
    __syncthreads();
    for (int off = 1; off < 256; off <<= 1) {
        int x = (t >= off) ? s[t - off] : 0;
        __syncthreads();
        s[t] += x;
        __syncthreads();
    }
    if (i < N) ptr[i] = s[t] - v;              // exclusive
    if (t == 255) blockSums[blockIdx.x] = s[255];
}

// Single-block exclusive scan of block sums (nb <= 256).
__global__ void scan_sums(int* __restrict__ blockSums, int nb) {
    __shared__ int s[256];
    int t = threadIdx.x;
    int v = (t < nb) ? blockSums[t] : 0;
    s[t] = v;
    __syncthreads();
    for (int off = 1; off < 256; off <<= 1) {
        int x = (t >= off) ? s[t - off] : 0;
        __syncthreads();
        s[t] += x;
        __syncthreads();
    }
    if (t < nb) blockSums[t] = s[t] - v;       // exclusive
}

__global__ void scan_add(int* __restrict__ ptr, const int* __restrict__ blockSums,
                         int* __restrict__ cursor, int N, int E) {
    int i = blockIdx.x * blockDim.x + threadIdx.x;
    if (i < N) {
        int p = ptr[i] + blockSums[i >> 8];
        ptr[i] = p;
        cursor[i] = p;
    }
    if (i == 0) ptr[N] = E;
}

// One int atomic per edge (vs 256 f32 atomics per edge before).
__global__ void csr_fill(const int* __restrict__ src, const int* __restrict__ dst,
                         int* __restrict__ cursor, int* __restrict__ eidx, int E) {
    int e = blockIdx.x * blockDim.x + threadIdx.x;
    if (e < E) {
        int pos = atomicAdd(&cursor[dst[e]], 1);
        eidx[pos] = src[e];
    }
}

// ------------------------------------------------------- mean aggregation
// One wave per destination row: 8 f32 accumulators/lane (256-wide row),
// neighbor rows stream in as coalesced float4 pairs; emits bf16 directly.
__global__ void aggregate_csr256(const float* __restrict__ X,
                                 const int* __restrict__ ptr,
                                 const int* __restrict__ eidx,
                                 ushort_t* __restrict__ aggbf, int N) {
    int row = blockIdx.x * blockDim.y + threadIdx.y;
    if (row >= N) return;
    const int l   = threadIdx.x;        // lane: cols l*8 .. l*8+7
    const int beg = ptr[row];
    const int end = ptr[row + 1];
    float a0=0,a1=0,a2=0,a3=0,a4=0,a5=0,a6=0,a7=0;
    for (int e = beg; e < end; ++e) {
        const float4* r4 = reinterpret_cast<const float4*>(
            X + (size_t)eidx[e] * 256 + l * 8);
        float4 x = r4[0], y = r4[1];
        a0+=x.x; a1+=x.y; a2+=x.z; a3+=x.w;
        a4+=y.x; a5+=y.y; a6+=y.z; a7+=y.w;
    }
    float inv = (end > beg) ? 1.0f / (float)(end - beg) : 0.0f;
    union { uint4 q; ushort_t h[8]; } o;
    o.h[0]=f32_to_bf16(a0*inv); o.h[1]=f32_to_bf16(a1*inv);
    o.h[2]=f32_to_bf16(a2*inv); o.h[3]=f32_to_bf16(a3*inv);
    o.h[4]=f32_to_bf16(a4*inv); o.h[5]=f32_to_bf16(a5*inv);
    o.h[6]=f32_to_bf16(a6*inv); o.h[7]=f32_to_bf16(a7*inv);
    *reinterpret_cast<uint4*>(aggbf + (size_t)row * 256 + l * 8) = o.q;
}

// ---------------------------------------------------------------- converts
__global__ void f32_to_bf16_kernel(const float* __restrict__ in,
                                   ushort_t* __restrict__ out, int n) {
    int i = blockIdx.x * blockDim.x + threadIdx.x;
    if (i < n) out[i] = f32_to_bf16(in[i]);
}

__global__ void wconv256(const float* __restrict__ W, ushort_t* __restrict__ Wbf,
                         int Dout, int DoutPad) {
    int i = blockIdx.x * blockDim.x + threadIdx.x;
    if (i >= DoutPad * 256) return;
    int o = i >> 8;
    Wbf[i] = f32_to_bf16(o < Dout ? W[i] : 0.0f);
}

// ---------------------------------------------------------------- WMMA GEMM
// Y = Xbf*Wself^T + Aggbf*Wneigh^T + bias.
// Each wave: 64x16 output (4 M-tiles), B fragments reused across 8 WMMAs/K-step.
__global__ void sage_gemm_wmma(const ushort_t* __restrict__ Xbf,
                               const ushort_t* __restrict__ Abf,
                               const ushort_t* __restrict__ WsBf,
                               const ushort_t* __restrict__ WnBf,
                               const float* __restrict__ bias,
                               float* __restrict__ Y,
                               int Din, int Dout, int ldY, int tilesM) {
    const int mt0 = (blockIdx.x * blockDim.y + threadIdx.y) * 4;
    if (mt0 >= tilesM) return;                  // wave-uniform exit
    int nmt = tilesM - mt0; if (nmt > 4) nmt = 4;
    const int ntile = blockIdx.y;
    const int lane  = threadIdx.x;
    const int half  = lane >> 4;
    const int l15   = lane & 15;
    const int kb    = half * 8;

    const ushort_t* bs = WsBf + (size_t)(ntile * 16 + l15) * Din;
    const ushort_t* bn = WnBf + (size_t)(ntile * 16 + l15) * Din;
    const size_t rowStride = (size_t)16 * Din;
    const ushort_t* ax0 = Xbf + (size_t)(mt0 * 16 + l15) * Din;
    const ushort_t* aa0 = Abf + (size_t)(mt0 * 16 + l15) * Din;

    v8f c0 = {}, c1 = {}, c2 = {}, c3 = {};

    if (nmt == 4) {                             // hot path: full 64-row block
        for (int k0 = 0; k0 < Din; k0 += 32) {
            const int ko = k0 + kb;
            v16bf fbS = load_frag(bs + ko);
            v16bf fbN = load_frag(bn + ko);
            c0 = wmma_bf16(load_frag(ax0 + ko), fbS, c0);
            c0 = wmma_bf16(load_frag(aa0 + ko), fbN, c0);
            c1 = wmma_bf16(load_frag(ax0 + rowStride + ko), fbS, c1);
            c1 = wmma_bf16(load_frag(aa0 + rowStride + ko), fbN, c1);
            c2 = wmma_bf16(load_frag(ax0 + 2 * rowStride + ko), fbS, c2);
            c2 = wmma_bf16(load_frag(aa0 + 2 * rowStride + ko), fbN, c2);
            c3 = wmma_bf16(load_frag(ax0 + 3 * rowStride + ko), fbS, c3);
            c3 = wmma_bf16(load_frag(aa0 + 3 * rowStride + ko), fbN, c3);
        }
    } else {                                    // remainder M-tiles (uniform nmt)
        for (int k0 = 0; k0 < Din; k0 += 32) {
            const int ko = k0 + kb;
            v16bf fbS = load_frag(bs + ko);
            v16bf fbN = load_frag(bn + ko);
            c0 = wmma_bf16(load_frag(ax0 + ko), fbS, c0);
            c0 = wmma_bf16(load_frag(aa0 + ko), fbN, c0);
            if (nmt > 1) {
                c1 = wmma_bf16(load_frag(ax0 + rowStride + ko), fbS, c1);
                c1 = wmma_bf16(load_frag(aa0 + rowStride + ko), fbN, c1);
            }
            if (nmt > 2) {
                c2 = wmma_bf16(load_frag(ax0 + 2 * rowStride + ko), fbS, c2);
                c2 = wmma_bf16(load_frag(aa0 + 2 * rowStride + ko), fbN, c2);
            }
        }
    }

    const int ncol = ntile * 16 + l15;
    if (ncol < Dout) {
        const float bv = bias[ncol];
        v8f acc[4] = { c0, c1, c2, c3 };
        for (int t = 0; t < nmt; ++t) {
            #pragma unroll
            for (int r = 0; r < 8; ++r) {
                int mrow = (mt0 + t) * 16 + r + half * 8;  // C/D VGPR layout
                Y[(size_t)mrow * ldY + ncol] = acc[t][r] + bv;
            }
        }
    }
}

// ---------------------------------------------------------------- batchnorm
__global__ void bn_stats256(const float* __restrict__ H, int Nrows,
                            float* __restrict__ sum, float* __restrict__ sumsq) {
    int col = threadIdx.x;
    int r0 = blockIdx.x * 128;
    int r1 = r0 + 128; if (r1 > Nrows) r1 = Nrows;
    float s = 0.f, s2 = 0.f;
    for (int r = r0; r < r1; ++r) {
        float v = H[(size_t)r * 256 + col];
        s += v; s2 += v * v;
    }
    atomicAdd(&sum[col], s);
    atomicAdd(&sumsq[col], s2);
}

__global__ void bn_finalize256(const float* sum, const float* sumsq,
                               const float* w, const float* b, float invN,
                               float* scale, float* shift) {
    int c = threadIdx.x;
    float m  = sum[c] * invN;
    float v  = sumsq[c] * invN - m * m;
    float sc = w[c] * rsqrtf(v + 1e-5f);
    scale[c] = sc;
    shift[c] = b[c] - m * sc;
}

__global__ void apply_bn_relu_res256(float* __restrict__ H,
                                     const float* __restrict__ R,
                                     const float* __restrict__ scA, const float* __restrict__ shA,
                                     const float* __restrict__ scB, const float* __restrict__ shB,
                                     ushort_t* __restrict__ Hbf, int n) {
    int i = blockIdx.x * blockDim.x + threadIdx.x;
    if (i >= n) return;
    int c = i & 255;
    float v = H[i] * scA[c] + shA[c] + R[i] * scB[c] + shB[c];
    v = fmaxf(v, 0.0f);
    H[i] = v;
    Hbf[i] = f32_to_bf16(v);
}

__global__ void apply_bn_relu256(float* __restrict__ H,
                                 const float* __restrict__ sc, const float* __restrict__ sh,
                                 ushort_t* __restrict__ Hbf, int n) {
    int i = blockIdx.x * blockDim.x + threadIdx.x;
    if (i >= n) return;
    int c = i & 255;
    float v = fmaxf(H[i] * sc[c] + sh[c], 0.0f);
    H[i] = v;
    Hbf[i] = f32_to_bf16(v);
}

// ---------------------------------------------------------------- launch
extern "C" void kernel_launch(void* const* d_in, const int* in_sizes, int n_in,
                              void* d_out, int out_size, void* d_ws, size_t ws_size,
                              hipStream_t stream) {
    const float* LLM = (const float*)d_in[0];
    const float* PLM = (const float*)d_in[1];
    const int*   src = (const int*)d_in[2];
    const int*   dst = (const int*)d_in[3];
    const float* Ws0 = (const float*)d_in[4];
    const float* Wn0 = (const float*)d_in[5];
    const float* B0  = (const float*)d_in[6];
    const float* Ws1 = (const float*)d_in[7];
    const float* Wn1 = (const float*)d_in[8];
    const float* B1  = (const float*)d_in[9];
    const float* Ws2 = (const float*)d_in[10];
    const float* Wn2 = (const float*)d_in[11];
    const float* B2  = (const float*)d_in[12];
    const float* bn0w = (const float*)d_in[13];
    const float* bn0b = (const float*)d_in[14];
    const float* bn1w = (const float*)d_in[15];
    const float* bn1b = (const float*)d_in[16];

    const int D  = 256;
    const int N  = in_sizes[0] / D;         // 50000
    const int E  = in_sizes[2];             // 800000
    const int ND = N * D;                   // 12.8M
    const int NCLS = 40, NCLS_PAD = 48;

    // ---- carve workspace
    char* p = (char*)d_ws;
    auto carve = [&](size_t bytes) { void* r = p; p += (bytes + 255) & ~(size_t)255; return r; };
    int*      deg   = (int*)carve((size_t)N * 4);
    int*      ptr   = (int*)carve((size_t)(N + 1) * 4);
    int*      cur   = (int*)carve((size_t)N * 4);
    int*      eidx  = (int*)carve((size_t)E * 4);
    int*      bsums = (int*)carve(256 * 4);
    float*    hA    = (float*)carve((size_t)ND * 4);
    float*    hB    = (float*)carve((size_t)ND * 4);
    ushort_t* Xbf   = (ushort_t*)carve((size_t)ND * 2);
    ushort_t* Abf   = (ushort_t*)carve((size_t)ND * 2);
    ushort_t* WsBf  = (ushort_t*)carve((size_t)256 * 256 * 2);
    ushort_t* WnBf  = (ushort_t*)carve((size_t)256 * 256 * 2);
    float*    sumA  = (float*)carve(256 * 4);
    float*    sqA   = (float*)carve(256 * 4);
    float*    sumB  = (float*)carve(256 * 4);
    float*    sqB   = (float*)carve(256 * 4);
    float*    scA   = (float*)carve(256 * 4);
    float*    shA   = (float*)carve(256 * 4);
    float*    scB   = (float*)carve(256 * 4);
    float*    shB   = (float*)carve(256 * 4);

    const int TB = 256;
    const int tilesM = (N + 15) / 16;       // 3125
    const int nScanBlk = (N + 255) / 256;   // 196 (<=256)
    const float invN = 1.0f / (float)N;
    dim3 gblk(32, 4);
    dim3 ggrid((((tilesM + 3) / 4) + 3) / 4, 16);
    dim3 agBlk(32, 8);
    dim3 agGrid((N + 7) / 8);

    // ---- build CSR once (graph identical for all 3 layers)
    fill_zero_i32<<<(N + TB - 1) / TB, TB, 0, stream>>>(deg, N);
    deg_count<<<(E + TB - 1) / TB, TB, 0, stream>>>(dst, deg, E);
    scan_block<<<nScanBlk, 256, 0, stream>>>(deg, ptr, bsums, N);
    scan_sums<<<1, 256, 0, stream>>>(bsums, nScanBlk);
    scan_add<<<(N + TB - 1) / TB, TB, 0, stream>>>(ptr, bsums, cur, N, E);
    csr_fill<<<(E + TB - 1) / TB, TB, 0, stream>>>(src, dst, cur, eidx, E);

    // ================= layer 0 =================
    aggregate_csr256<<<agGrid, agBlk, 0, stream>>>(LLM, ptr, eidx, Abf, N);
    f32_to_bf16_kernel<<<(ND + TB - 1) / TB, TB, 0, stream>>>(LLM, Xbf, ND);
    wconv256<<<(256 * 256 + TB - 1) / TB, TB, 0, stream>>>(Ws0, WsBf, 256, 256);
    wconv256<<<(256 * 256 + TB - 1) / TB, TB, 0, stream>>>(Wn0, WnBf, 256, 256);
    sage_gemm_wmma<<<ggrid, gblk, 0, stream>>>(Xbf, Abf, WsBf, WnBf, B0, hA,
                                               256, 256, 256, tilesM);
    fill_zero_f32<<<4, 256, 0, stream>>>(sumA, 1024);   // sumA..sqB contiguous
    bn_stats256<<<(N + 127) / 128, 256, 0, stream>>>(hA, N, sumA, sqA);
    bn_stats256<<<(N + 127) / 128, 256, 0, stream>>>(PLM, N, sumB, sqB);
    bn_finalize256<<<1, 256, 0, stream>>>(sumA, sqA, bn0w, bn0b, invN, scA, shA);
    bn_finalize256<<<1, 256, 0, stream>>>(sumB, sqB, bn0w, bn0b, invN, scB, shB);
    apply_bn_relu_res256<<<(ND + TB - 1) / TB, TB, 0, stream>>>(
        hA, PLM, scA, shA, scB, shB, Xbf, ND);

    // ================= layer 1 =================
    aggregate_csr256<<<agGrid, agBlk, 0, stream>>>(hA, ptr, eidx, Abf, N);
    wconv256<<<(256 * 256 + TB - 1) / TB, TB, 0, stream>>>(Ws1, WsBf, 256, 256);
    wconv256<<<(256 * 256 + TB - 1) / TB, TB, 0, stream>>>(Wn1, WnBf, 256, 256);
    sage_gemm_wmma<<<ggrid, gblk, 0, stream>>>(Xbf, Abf, WsBf, WnBf, B1, hB,
                                               256, 256, 256, tilesM);
    fill_zero_f32<<<4, 256, 0, stream>>>(sumA, 1024);
    bn_stats256<<<(N + 127) / 128, 256, 0, stream>>>(hB, N, sumA, sqA);
    bn_finalize256<<<1, 256, 0, stream>>>(sumA, sqA, bn1w, bn1b, invN, scA, shA);
    apply_bn_relu256<<<(ND + TB - 1) / TB, TB, 0, stream>>>(hB, scA, shA, Xbf, ND);

    // ================= layer 2 (output) =================
    aggregate_csr256<<<agGrid, agBlk, 0, stream>>>(hB, ptr, eidx, Abf, N);
    wconv256<<<(NCLS_PAD * 256 + TB - 1) / TB, TB, 0, stream>>>(Ws2, WsBf, NCLS, NCLS_PAD);
    wconv256<<<(NCLS_PAD * 256 + TB - 1) / TB, TB, 0, stream>>>(Wn2, WnBf, NCLS, NCLS_PAD);
    sage_gemm_wmma<<<dim3(ggrid.x, NCLS_PAD / 16), gblk, 0, stream>>>(
        Xbf, Abf, WsBf, WnBf, B2, (float*)d_out, 256, NCLS, NCLS, tilesM);
}
